// CausalSelfAttentionRotary_23991687315662
// MI455X (gfx1250) — matrix operand order repaired and checked by reading, not compiled, not run on hardware
//
#include <hip/hip_runtime.h>
#include <hip/hip_bf16.h>
#include <stdint.h>

#define D_MODEL 1024
#define N_HEADS 16
#define D_K     64
#define SEQ_T   2048
#define BATCH   2
#define NTOK    (BATCH * SEQ_T)   // 4096 tokens

typedef __bf16 bf16;
typedef __attribute__((ext_vector_type(16))) __bf16   v16bf;
typedef __attribute__((ext_vector_type(8)))  float    v8f;
typedef __attribute__((ext_vector_type(4)))  uint32_t v4u;

union Frag { v16bf v; v4u q[2]; };

__device__ __forceinline__ v8f wmma_bf16(v16bf a, v16bf b, v8f c) {
  // D = A(16x32 bf16) * B(32x16 bf16) + C(16x16 f32)
  return __builtin_amdgcn_wmma_f32_16x16x32_bf16(
      /*neg_a=*/false, a, /*neg_b=*/false, b,
      /*c_mod=*/(short)0, c, /*reuse_a=*/false, /*reuse_b=*/false);
}

// ---------------------------------------------------------------------------
// DPP16 butterfly reductions across each 16-lane group (wave32 "row").
// VALU-only: no LDS round trip, unlike __shfl_xor -> ds_bpermute.
//   0xB1  = quad_perm [1,0,3,2]  (xor 1)
//   0x4E  = quad_perm [2,3,0,1]  (xor 2)
//   0x141 = row_half_mirror      (pairs the two quads of each 8-group)
//   0x140 = row_mirror           (pairs the two 8-groups of the row)
// ---------------------------------------------------------------------------
template <int CTRL>
__device__ __forceinline__ float dpp_f32(float x) {
  int xi = __float_as_int(x);
  int r  = __builtin_amdgcn_update_dpp(xi, xi, CTRL, 0xF, 0xF, false);
  return __int_as_float(r);
}
__device__ __forceinline__ float red_max16(float x) {
  x = fmaxf(x, dpp_f32<0xB1>(x));
  x = fmaxf(x, dpp_f32<0x4E>(x));
  x = fmaxf(x, dpp_f32<0x141>(x));
  x = fmaxf(x, dpp_f32<0x140>(x));
  return x;
}
__device__ __forceinline__ float red_sum16(float x) {
  x += dpp_f32<0xB1>(x);
  x += dpp_f32<0x4E>(x);
  x += dpp_f32<0x141>(x);
  x += dpp_f32<0x140>(x);
  return x;
}

// ---------------------------------------------------------------------------
// f32 -> bf16 conversion (x, Wqkv, Wout)
// ---------------------------------------------------------------------------
__global__ __launch_bounds__(256) void f32_to_bf16_kernel(
    const float* __restrict__ in, bf16* __restrict__ out, int n) {
  int i = blockIdx.x * 256 + threadIdx.x;
  if (i < n) out[i] = (bf16)in[i];
}

// ---------------------------------------------------------------------------
// Generic bf16 GEMM:  out[m,n] = sum_k A[m,k] * Bt[n,k] + bias[n]
// A: M x K row-major bf16, Bt: N x K row-major bf16 (computes A @ Bt^T).
// Block: 256 threads = 8 waves (4 along M x 2 along N). Wave tile: 32x64
// (2 A-frags x 4 B-frags -> 8 WMMAs per K-step, 12 b128 loads).
// Block tile: 128(M) x 128(N). Requires M%128==0, N%128==0, K%32==0.
//
// Fragment addressing (CDNA5 ISA 7.12.2, wave32):
//  A-frag (16x32): lane m = lane&15, half = lane>>4;
//      dwords 0..3 = K[half*8 + 0..7], dwords 4..7 = K[16 + half*8 + 0..7].
//  B-frag (32x16): lane n = lane&15, half = lane>>4;
//      dwords 0..3 = K[half*8 + 0..7], dwords 4..7 = K[16 + half*8 + 0..7] (as A).
// ---------------------------------------------------------------------------
template <bool OUT_BF16>
__global__ __launch_bounds__(256) void gemm_bf16_kernel(
    const bf16* __restrict__ A, const bf16* __restrict__ Bt,
    const float* __restrict__ bias, void* __restrict__ outp,
    int M, int N, int K) {
  const int lane = threadIdx.x & 31;
  const int wave = threadIdx.x >> 5;
  const int wm   = wave & 3;
  const int wn   = wave >> 2;
  const int m0   = blockIdx.y * 128 + wm * 32;
  const int n0   = blockIdx.x * 128 + wn * 64;
  const int col  = lane & 15;
  const int half = lane >> 4;
  const int ha   = half * 8;

  v8f acc[2][4] = {};

  const bf16* a0 = A  + (size_t)(m0 + col) * K;
  const bf16* a1 = a0 + (size_t)16 * K;
  const bf16* bbase = Bt + (size_t)(n0 + col) * K + half * 8;

  for (int kk = 0; kk < K; kk += 32) {
    __builtin_prefetch(a0 + kk + 256, 0, 0);
    __builtin_prefetch(bbase + kk + 256, 0, 0);
    Frag af0, af1, bfr[4];
    af0.q[0] = *(const v4u*)(a0 + kk + ha);
    af0.q[1] = *(const v4u*)(a0 + kk + 16 + ha);
    af1.q[0] = *(const v4u*)(a1 + kk + ha);
    af1.q[1] = *(const v4u*)(a1 + kk + 16 + ha);
#pragma unroll
    for (int j = 0; j < 4; ++j) {
      const bf16* bp = bbase + (size_t)(j * 16) * K + kk;
      bfr[j].q[0] = *(const v4u*)(bp);
      bfr[j].q[1] = *(const v4u*)(bp + 16);
    }
#pragma unroll
    for (int j = 0; j < 4; ++j) {
      acc[0][j] = wmma_bf16(af0.v, bfr[j].v, acc[0][j]);
      acc[1][j] = wmma_bf16(af1.v, bfr[j].v, acc[1][j]);
    }
  }

  // Epilogue: C/D layout -> lane holds col n = lane&15, rows r + 8*half.
#pragma unroll
  for (int mt = 0; mt < 2; ++mt) {
#pragma unroll
    for (int nt = 0; nt < 4; ++nt) {
      const int n = n0 + nt * 16 + col;
      const float bv = bias ? bias[n] : 0.0f;
#pragma unroll
      for (int r = 0; r < 8; ++r) {
        const int m = m0 + mt * 16 + r + half * 8;
        const float val = acc[mt][nt][r] + bv;
        if (OUT_BF16) ((bf16*)outp)[(size_t)m * N + n] = (bf16)val;
        else          ((float*)outp)[(size_t)m * N + n] = val;
      }
    }
  }
}

// ---------------------------------------------------------------------------
// RoPE + head split.  qkv: [NTOK][3072] bf16 (per head: q[64] k[64] v[64]).
// Writes q (rotated, PRE-SCALED by 1/sqrt(d_k)) and k (rotated) as
// [bh][t][64] bf16, and v TRANSPOSED as [bh][64][t] bf16 so the P@V
// B-fragments are contiguous global loads in the attention kernel.
// One thread per (token, head, rotation pair): 4096*16*32 threads.
// ---------------------------------------------------------------------------
__global__ __launch_bounds__(256) void rope_split_kernel(
    const bf16* __restrict__ qkv, bf16* __restrict__ qd,
    bf16* __restrict__ kd, bf16* __restrict__ vtd) {
  const int idx = blockIdx.x * 256 + threadIdx.x;
  const int i = idx & 31;          // rotation pair 0..31
  const int h = (idx >> 5) & 15;   // head
  const int t = idx >> 9;          // global token 0..4095
  if (t >= NTOK) return;
  const int b  = t >> 11;
  const int tt = t & (SEQ_T - 1);

  const bf16* src = qkv + (size_t)t * (3 * D_MODEL) + h * (3 * D_K);
  const float q1 = (float)src[2 * i],        q2 = (float)src[2 * i + 1];
  const float k1 = (float)src[D_K + 2 * i],  k2 = (float)src[D_K + 2 * i + 1];

  const float freq = __powf(10000.0f, -(float)(2 * i) / (float)D_K);
  float s, c;
  __sincosf((float)tt * freq, &s, &c);

  const float qscale = 0.125f;     // 1/sqrt(64), folded into Q
  const size_t bh = (size_t)(b * N_HEADS + h);
  bf16* qo = qd + bh * SEQ_T * D_K + (size_t)tt * D_K;
  bf16* ko = kd + bh * SEQ_T * D_K + (size_t)tt * D_K;
  qo[2 * i]     = (bf16)((q1 * c - q2 * s) * qscale);
  qo[2 * i + 1] = (bf16)((q1 * s + q2 * c) * qscale);
  ko[2 * i]     = (bf16)(k1 * c - k2 * s);
  ko[2 * i + 1] = (bf16)(k1 * s + k2 * c);

  bf16* vo = vtd + bh * (size_t)D_K * SEQ_T + tt;
  vo[(size_t)(2 * i) * SEQ_T]     = src[2 * D_K + 2 * i];
  vo[(size_t)(2 * i + 1) * SEQ_T] = src[2 * D_K + 2 * i + 1];
}

// ---------------------------------------------------------------------------
// Causal flash attention, one (b,h) per blockIdx.y, 64 q rows per block.
// 128 threads = 4 waves; each wave owns a 16-row q tile, iterates keys in
// steps of 32 with online softmax. Key loop is split into an UNMASKED fast
// path (all key blocks strictly below the diagonal) plus one masked tail
// block. Row-wise softmax reductions use DPP16 butterflies (VALU-only).
// S/P transposed C-layout -> A-layout via per-wave LDS tile.
// attn out: [token][h*64+dk] bf16 (A matrix of the output projection GEMM).
// ---------------------------------------------------------------------------
__global__ __launch_bounds__(128) void flash_attn_kernel(
    const bf16* __restrict__ Q, const bf16* __restrict__ Kmat,
    const bf16* __restrict__ Vt, bf16* __restrict__ attn) {
  __shared__ float sS[4][16][34];   // per-wave 16x32 P scratch (+pad)

  const int lane = threadIdx.x & 31;
  const int wave = threadIdx.x >> 5;
  const int col  = lane & 15;
  const int half = lane >> 4;
  const int bh = blockIdx.y;
  const int b  = bh >> 4;
  const int h  = bh & 15;
  const int q0 = blockIdx.x * 64 + wave * 16;   // token within sequence

  const bf16* qb = Q    + (size_t)bh * SEQ_T * D_K;
  const bf16* kb = Kmat + (size_t)bh * SEQ_T * D_K;
  const bf16* vb = Vt   + (size_t)bh * D_K * SEQ_T;

  // Q fragments (A-layout), held in registers for the whole key loop.
  Frag qf0, qf1;
  {
    const bf16* qp = qb + (size_t)(q0 + col) * D_K;
    qf0.q[0] = *(const v4u*)(qp + half * 16);
    qf0.q[1] = *(const v4u*)(qp + 8 + half * 16);
    qf1.q[0] = *(const v4u*)(qp + 32 + half * 16);
    qf1.q[1] = *(const v4u*)(qp + 40 + half * 16);
  }

  v8f o0 = {}, o1 = {}, o2 = {}, o3 = {};   // O accum: 16 x 64 (4 dk tiles)
  float mrow[8], lrow[8];
#pragma unroll
  for (int r = 0; r < 8; ++r) { mrow[r] = -3.0e38f; lrow[r] = 0.0f; }

  float (*sP)[34] = sS[wave];

  // One key-block step; `masked` is a compile-time constant at each call
  // site after inlining, so the fast path carries no cmp/cndmask ops.
  auto step = [&](int kk0, bool masked) {
    // ---- S = Q @ K^T for 32 keys (two 16x16 tiles, 4 WMMAs) ----
    v8f s0 = {}, s1 = {};
    {
      const bf16* kp0 = kb + (size_t)(kk0 + col) * D_K + half * 16;
      const bf16* kp1 = kp0 + 16 * D_K;
      Frag kf;
      kf.q[0] = *(const v4u*)(kp0);       kf.q[1] = *(const v4u*)(kp0 + 8);
      s0 = wmma_bf16(qf0.v, kf.v, s0);
      kf.q[0] = *(const v4u*)(kp0 + 32);  kf.q[1] = *(const v4u*)(kp0 + 40);
      s0 = wmma_bf16(qf1.v, kf.v, s0);
      kf.q[0] = *(const v4u*)(kp1);       kf.q[1] = *(const v4u*)(kp1 + 8);
      s1 = wmma_bf16(qf0.v, kf.v, s1);
      kf.q[0] = *(const v4u*)(kp1 + 32);  kf.q[1] = *(const v4u*)(kp1 + 40);
      s1 = wmma_bf16(qf1.v, kf.v, s1);
    }

    // ---- (mask +) online softmax; DPP16 row reductions ----
#pragma unroll
    for (int r = 0; r < 8; ++r) {
      float a0 = s0[r], a1 = s1[r];
      if (masked) {
        const int qi = q0 + r + half * 8;
        const float NEG = -3.0e38f;
        a0 = (kk0 + col      <= qi) ? a0 : NEG;
        a1 = (kk0 + 16 + col <= qi) ? a1 : NEG;
      }
      const float mx    = red_max16(fmaxf(a0, a1));
      const float mnew  = fmaxf(mrow[r], mx);
      const float alpha = __expf(mrow[r] - mnew);
      const float p0 = __expf(a0 - mnew);
      const float p1 = __expf(a1 - mnew);
      const float sum = red_sum16(p0 + p1);
      lrow[r] = lrow[r] * alpha + sum;
      mrow[r] = mnew;
      o0[r] *= alpha; o1[r] *= alpha; o2[r] *= alpha; o3[r] *= alpha;
      sP[r + half * 8][col]      = p0;   // C-layout -> LDS (row-major 16x32)
      sP[r + half * 8][col + 16] = p1;
    }
    asm volatile("s_wait_dscnt 0" ::: "memory");  // LDS stores -> reads

    // ---- P: LDS -> A-layout bf16 fragment ----
    Frag pf;
    {
      const float* prow = sP[col];   // A-frag: lane holds row m = lane&15
#pragma unroll
      for (int e = 0; e < 16; ++e) {
        const int kx = half * 16 + e;
        pf.v[e] = (bf16)prow[kx];
      }
    }

    // ---- O += P @ V  (V pre-transposed: contiguous B-frag loads) ----
    {
      Frag vf;
      const bf16* vp = vb + (size_t)col * SEQ_T + kk0 + half * 16;
      vf.q[0] = *(const v4u*)(vp);     vf.q[1] = *(const v4u*)(vp + 8);
      o0 = wmma_bf16(pf.v, vf.v, o0);
      vp += (size_t)16 * SEQ_T;
      vf.q[0] = *(const v4u*)(vp);     vf.q[1] = *(const v4u*)(vp + 8);
      o1 = wmma_bf16(pf.v, vf.v, o1);
      vp += (size_t)16 * SEQ_T;
      vf.q[0] = *(const v4u*)(vp);     vf.q[1] = *(const v4u*)(vp + 8);
      o2 = wmma_bf16(pf.v, vf.v, o2);
      vp += (size_t)16 * SEQ_T;
      vf.q[0] = *(const v4u*)(vp);     vf.q[1] = *(const v4u*)(vp + 8);
      o3 = wmma_bf16(pf.v, vf.v, o3);
    }
  };

  // Fast path: key blocks entirely below the diagonal (no masking needed:
  // max key = kk0+31 < q0 <= every row index of this q-tile).
  int kk0 = 0;
  for (; kk0 + 32 <= q0; kk0 += 32) step(kk0, false);
  // Exactly one masked tail block covers keys [kk0, q0+16).
  for (; kk0 < q0 + 16; kk0 += 32) step(kk0, true);

  // ---- normalize + store attn[token][h*64 + dk] ----
  const size_t obase = ((size_t)b * SEQ_T + q0) * D_MODEL + (size_t)h * D_K;
#pragma unroll
  for (int r = 0; r < 8; ++r) {
    const float inv = 1.0f / lrow[r];
    const size_t ro = obase + (size_t)(r + half * 8) * D_MODEL;
    attn[ro +  0 + col] = (bf16)(o0[r] * inv);
    attn[ro + 16 + col] = (bf16)(o1[r] * inv);
    attn[ro + 32 + col] = (bf16)(o2[r] * inv);
    attn[ro + 48 + col] = (bf16)(o3[r] * inv);
  }
}

// ---------------------------------------------------------------------------
extern "C" void kernel_launch(void* const* d_in, const int* in_sizes, int n_in,
                              void* d_out, int out_size, void* d_ws, size_t ws_size,
                              hipStream_t stream) {
  (void)in_sizes; (void)n_in; (void)out_size; (void)ws_size;
  const float* x    = (const float*)d_in[0];   // [4096,1024]
  const float* Wqkv = (const float*)d_in[1];   // [3072,1024]
  const float* bqkv = (const float*)d_in[2];   // [3072]
  const float* Wout = (const float*)d_in[3];   // [1024,1024]
  const float* bout = (const float*)d_in[4];   // [1024]
  float* out = (float*)d_out;                  // [4096,1024] f32

  // Workspace bump allocation (~72 MB total).
  char* p = (char*)d_ws;
  bf16* xb    = (bf16*)p; p += (size_t)NTOK * D_MODEL * 2;          //  8 MB
  bf16* wqkvb = (bf16*)p; p += (size_t)3 * D_MODEL * D_MODEL * 2;   //  6 MB
  bf16* woutb = (bf16*)p; p += (size_t)D_MODEL * D_MODEL * 2;       //  2 MB
  bf16* qkvb  = (bf16*)p; p += (size_t)NTOK * 3 * D_MODEL * 2;      // 24 MB
  const size_t headElems = (size_t)BATCH * N_HEADS * SEQ_T * D_K;   // 4.19M
  bf16* qbuf  = (bf16*)p; p += headElems * 2;                       //  8 MB
  bf16* kbuf  = (bf16*)p; p += headElems * 2;                       //  8 MB
  bf16* vtbuf = (bf16*)p; p += headElems * 2;                       //  8 MB
  bf16* attnb = (bf16*)p; p += (size_t)NTOK * D_MODEL * 2;          //  8 MB

  // 1) Down-convert operands to bf16.
  {
    const int nx = NTOK * D_MODEL;
    const int nw = 3 * D_MODEL * D_MODEL;
    const int no = D_MODEL * D_MODEL;
    f32_to_bf16_kernel<<<(nx + 255) / 256, 256, 0, stream>>>(x, xb, nx);
    f32_to_bf16_kernel<<<(nw + 255) / 256, 256, 0, stream>>>(Wqkv, wqkvb, nw);
    f32_to_bf16_kernel<<<(no + 255) / 256, 256, 0, stream>>>(Wout, woutb, no);
  }

  // 2) QKV projection: qkv = x @ Wqkv^T + bqkv   (bf16 out)
  gemm_bf16_kernel<true><<<dim3((3 * D_MODEL) / 128, NTOK / 128), 256, 0, stream>>>(
      xb, wqkvb, bqkv, qkvb, NTOK, 3 * D_MODEL, D_MODEL);

  // 3) RoPE + per-head split (Q pre-scaled, V transposed).
  rope_split_kernel<<<(NTOK * N_HEADS * 32) / 256, 256, 0, stream>>>(
      qkvb, qbuf, kbuf, vtbuf);

  // 4) Causal flash attention.
  flash_attn_kernel<<<dim3(SEQ_T / 64, BATCH * N_HEADS), 128, 0, stream>>>(
      qbuf, kbuf, vtbuf, attnb);

  // 5) Output projection: out = attn @ Wout^T + bout   (f32 out)
  gemm_bf16_kernel<false><<<dim3(D_MODEL / 128, NTOK / 128), 256, 0, stream>>>(
      attnb, woutb, bout, out, NTOK, D_MODEL, D_MODEL);
}
